// GRNNTransformGated_670014898787
// MI455X (gfx1250) — matrix operand order repaired and checked by reading, not compiled, not run on hardware
//
#include <hip/hip_runtime.h>

#define NN   65536
#define FEAT 7
#define HID  64

typedef __attribute__((ext_vector_type(16))) __bf16       v16bf;
typedef __attribute__((ext_vector_type(8)))  float        v8f;
typedef __attribute__((ext_vector_type(4)))  unsigned int v4u;

__device__ __forceinline__ unsigned int f2bf1(float x) {
  union { float f; unsigned u; } v; v.f = x;
  unsigned r = v.u + 0x7FFFu + ((v.u >> 16) & 1u);   // round-to-nearest-even
  return r >> 16;
}
__device__ __forceinline__ float bf2f(unsigned short h) {
  union { unsigned u; float f; } v; v.u = ((unsigned)h) << 16;
  return v.f;
}

union FragCvt { struct { v4u lo, hi; } p; v16bf v; };

// 16-bit A-matrix 16x32 fragment from row-major bf16 [16][K] (ISA 7.12.2):
// lanes 0-15: row=lane, K = kbase+[0..8) in v0..3, kbase+[16..24) in v4..7
// lanes 16-31: row=lane-16, K = kbase+[8..16) / kbase+[24..32)
__device__ __forceinline__ v16bf load_afrag(const unsigned short* base, int K,
                                            int kbase, int lane) {
  int m  = lane & 15;
  int k0 = kbase + ((lane & 16) ? 8 : 0);
  const unsigned short* p = base + m * K + k0;
  FragCvt c;
  c.p.lo = *(const v4u*)p;
  c.p.hi = *(const v4u*)(p + 16);
  return c.v;
}

// 16-bit B-matrix 32x16 fragment: column n = n0 + lane%16 = weight row (B=W^T),
// lanes 0-15 hold K=kbase+[0..8),[16..24); lanes 16-31 hold K=kbase+[8..16),[24..32)
__device__ __forceinline__ v16bf load_bfrag(const unsigned short* __restrict__ W,
                                            int K, int n0, int kbase, int lane) {
  int n  = n0 + (lane & 15);
  int k0 = kbase + ((lane & 16) ? 8 : 0);
  const unsigned short* p = W + (size_t)n * K + k0;
  FragCvt c;
  c.p.lo = *(const v4u*)p;
  c.p.hi = *(const v4u*)(p + 16);
  return c.v;
}

__device__ __forceinline__ v8f wmma_bf16(v16bf a, v16bf b, v8f c) {
  return __builtin_amdgcn_wmma_f32_16x16x32_bf16(false, a, false, b,
                                                 (short)0, c, false, false);
}

__device__ __forceinline__ void cvt8_store(const float* __restrict__ src,
                                           unsigned short* dst) {
  float4 a = *(const float4*)src;
  float4 b = *(const float4*)(src + 4);
  v4u p;
  p.x = f2bf1(a.x) | (f2bf1(a.y) << 16);
  p.y = f2bf1(a.z) | (f2bf1(a.w) << 16);
  p.z = f2bf1(b.x) | (f2bf1(b.y) << 16);
  p.w = f2bf1(b.z) | (f2bf1(b.w) << 16);
  *(v4u*)dst = p;
}

// reductions across the 16 lanes of each half-wave (xor masks < 16)
__device__ __forceinline__ float hmax16(float x) {
  x = fmaxf(x, __shfl_xor(x, 1, 32));
  x = fmaxf(x, __shfl_xor(x, 2, 32));
  x = fmaxf(x, __shfl_xor(x, 4, 32));
  x = fmaxf(x, __shfl_xor(x, 8, 32));
  return x;
}
__device__ __forceinline__ float hsum16(float x) {
  x += __shfl_xor(x, 1, 32);
  x += __shfl_xor(x, 2, 32);
  x += __shfl_xor(x, 4, 32);
  x += __shfl_xor(x, 8, 32);
  return x;
}

// ---------------------------------------------------------------------------
// one scan step: 4 waves/block, 16 nodes per wave
// ---------------------------------------------------------------------------
__global__ __launch_bounds__(128) void grnn_step(
    const float* __restrict__ cont,           // [NN][7]   contents[level]
    const int*   __restrict__ child,          // [NN][2]   children[level]
    const float* __restrict__ hprev,          // [NN][64]  up_{level+1}
    const unsigned short* __restrict__ WrB,   // bf16 [192][192]
    const unsigned short* __restrict__ WhB,   // bf16 [64][192]
    const unsigned short* __restrict__ WzB,   // bf16 [256][256]
    const float* __restrict__ Wu,             // [64][7]
    const float* __restrict__ bu, const float* __restrict__ br,
    const float* __restrict__ bh, const float* __restrict__ bz,
    float* __restrict__ hout)                 // [NN][64]
{
  __shared__ unsigned short s_hhu[4][16 * 192];   // [hL|hR|u] bf16 per wave
  __shared__ unsigned short s_rh [4][16 * 192];   // r * hhu  bf16
  __shared__ unsigned short s_hH [4][16 * 64];    // h_H      bf16
  __shared__ float s_Wu[HID * FEAT];
  __shared__ float s_bu[HID], s_br[192], s_bh[HID], s_bz[256];

  const int tid  = threadIdx.x;
  const int wv   = tid >> 5;
  const int lane = tid & 31;
  const int nlo  = lane & 15;
  const int hh   = lane >> 4;           // half-wave: 0 or 1

  for (int i = tid; i < HID * FEAT; i += 128) s_Wu[i] = Wu[i];
  if (tid < HID) { s_bu[tid] = bu[tid]; s_bh[tid] = bh[tid]; }
  for (int i = tid; i < 192; i += 128) s_br[i] = br[i];
  for (int i = tid; i < 256; i += 128) s_bz[i] = bz[i];
  __syncthreads();

  const int m0   = blockIdx.x * 64 + wv * 16;
  const int node = m0 + nlo;

  unsigned short* hhu = s_hhu[wv];
  unsigned short* rhB = s_rh[wv];
  unsigned short* hHB = s_hH[wv];

  // ---- build hhu = [h_L | h_R | u] (bf16, row-major [16][192]) ----
  {
    int2 ch = *(const int2*)(child + (size_t)node * 2);
    const float* srcL = hprev + (size_t)ch.x * HID + hh * 32;
    const float* srcR = hprev + (size_t)ch.y * HID + hh * 32;
    unsigned short* dL = hhu + nlo * 192 +       hh * 32;
    unsigned short* dR = hhu + nlo * 192 + 64  + hh * 32;
#pragma unroll
    for (int j = 0; j < 32; j += 8) cvt8_store(srcL + j, dL + j);
#pragma unroll
    for (int j = 0; j < 32; j += 8) cvt8_store(srcR + j, dR + j);

    float c[FEAT];
    const float* cp = cont + (size_t)node * FEAT;
#pragma unroll
    for (int j = 0; j < FEAT; j++) c[j] = cp[j];

    unsigned short* dU = hhu + nlo * 192 + 128 + hh * 32;
#pragma unroll
    for (int fb = 0; fb < 32; fb += 8) {
      unsigned int pk[4];
#pragma unroll
      for (int e = 0; e < 8; e += 2) {
        int f = hh * 32 + fb + e;
        float x0 = s_bu[f], x1 = s_bu[f + 1];
#pragma unroll
        for (int j = 0; j < FEAT; j++) {
          x0 += s_Wu[f * FEAT + j]       * c[j];
          x1 += s_Wu[(f + 1) * FEAT + j] * c[j];
        }
        x0 = fmaxf(x0, 0.f); x1 = fmaxf(x1, 0.f);
        pk[e >> 1] = f2bf1(x0) | (f2bf1(x1) << 16);
      }
      v4u p; p.x = pk[0]; p.y = pk[1]; p.z = pk[2]; p.w = pk[3];
      *(v4u*)(dU + fb) = p;
    }
  }
  __syncthreads();

  // ---- r = sigmoid(hhu @ Wr^T + br);  rh = r * hhu ----
  v16bf ah[6];
#pragma unroll
  for (int kt = 0; kt < 6; kt++) ah[kt] = load_afrag(hhu, 192, kt * 32, lane);

#pragma unroll
  for (int nt = 0; nt < 12; nt++) {
    float bias = s_br[nt * 16 + nlo];
    v8f acc;
#pragma unroll
    for (int i = 0; i < 8; i++) acc[i] = bias;
#pragma unroll
    for (int kt = 0; kt < 6; kt++)
      acc = wmma_bf16(ah[kt], load_bfrag(WrB, 192, nt * 16, kt * 32, lane), acc);
#pragma unroll
    for (int i = 0; i < 8; i++) {
      int row = hh * 8 + i;                 // C-layout: M = i (+8 for hi half)
      int col = nt * 16 + nlo;
      float hv = bf2f(hhu[row * 192 + col]);
      float rv = 1.f / (1.f + __expf(-acc[i]));
      rhB[row * 192 + col] = (unsigned short)f2bf1(rv * hv);
    }
  }

  // ---- h_H = relu(rh @ Wh^T + bh), kept resident in f32 C-layout ----
  v16bf ar[6];
#pragma unroll
  for (int kt = 0; kt < 6; kt++) ar[kt] = load_afrag(rhB, 192, kt * 32, lane);

  float hHacc[4][8];
#pragma unroll
  for (int nt = 0; nt < 4; nt++) {
    float bias = s_bh[nt * 16 + nlo];
    v8f acc;
#pragma unroll
    for (int i = 0; i < 8; i++) acc[i] = bias;
#pragma unroll
    for (int kt = 0; kt < 6; kt++)
      acc = wmma_bf16(ar[kt], load_bfrag(WhB, 192, nt * 16, kt * 32, lane), acc);
#pragma unroll
    for (int i = 0; i < 8; i++) {
      float v = fmaxf(acc[i], 0.f);
      hHacc[nt][i] = v;
      int row = hh * 8 + i;
      hHB[row * 64 + nt * 16 + nlo] = (unsigned short)f2bf1(v);
    }
  }

  // ---- z = [h_H, hhu] @ Wz^T + bz; softmax over the 64-feature axis per
  //      gate (reference: transpose->(N,64,4), softmax(axis=1)); combine ----
  v16bf az[8];
  az[0] = load_afrag(hHB, 64, 0,  lane);
  az[1] = load_afrag(hHB, 64, 32, lane);
#pragma unroll
  for (int kt = 0; kt < 6; kt++) az[2 + kt] = ah[kt];

  float outacc[4][8];
#pragma unroll
  for (int j = 0; j < 4; j++)
#pragma unroll
    for (int i = 0; i < 8; i++) outacc[j][i] = 0.f;

#pragma unroll
  for (int g = 0; g < 4; g++) {                // gate = column block g*64..
    float zv[4][8];
#pragma unroll
    for (int j = 0; j < 4; j++) {              // feature sub-tile j*16..
      int nt = g * 4 + j;
      float bias = s_bz[nt * 16 + nlo];
      v8f acc;
#pragma unroll
      for (int i = 0; i < 8; i++) acc[i] = bias;
#pragma unroll
      for (int kt = 0; kt < 8; kt++)
        acc = wmma_bf16(az[kt], load_bfrag(WzB, 256, nt * 16, kt * 32, lane), acc);
#pragma unroll
      for (int i = 0; i < 8; i++) zv[j][i] = acc[i];
    }
    // softmax over 64 features: in-lane over j, butterfly over 16 lanes
#pragma unroll
    for (int i = 0; i < 8; i++) {
      float mx = fmaxf(fmaxf(zv[0][i], zv[1][i]), fmaxf(zv[2][i], zv[3][i]));
      mx = hmax16(mx);
      float e[4];
      float ps = 0.f;
#pragma unroll
      for (int j = 0; j < 4; j++) { e[j] = __expf(zv[j][i] - mx); ps += e[j]; }
      float inv = 1.f / hsum16(ps);
      int row = hh * 8 + i;
#pragma unroll
      for (int j = 0; j < 4; j++) {
        float w = e[j] * inv;
        float src = (g == 0) ? hHacc[j][i]
                  : bf2f(hhu[row * 192 + (g - 1) * 64 + j * 16 + nlo]);
        outacc[j][i] += w * src;
      }
    }
  }

#pragma unroll
  for (int j = 0; j < 4; j++)
#pragma unroll
    for (int i = 0; i < 8; i++) {
      int row = hh * 8 + i;
      hout[(size_t)(m0 + row) * HID + j * 16 + nlo] = outacc[j][i];
    }
}

// ---------------------------------------------------------------------------
__global__ void leaf_embed(const float* __restrict__ c, const float* __restrict__ Wu,
                           const float* __restrict__ bu, float* __restrict__ out) {
  int idx = blockIdx.x * blockDim.x + threadIdx.x;   // over NN*64
  int m = idx >> 6, f = idx & 63;
  const float* cp = c + (size_t)m * FEAT;
  const float* w  = Wu + f * FEAT;
  float x = bu[f];
#pragma unroll
  for (int j = 0; j < FEAT; j++) x += w[j] * cp[j];
  out[idx] = fmaxf(x, 0.f);
}

__global__ void wcvt(const float* __restrict__ src, unsigned short* __restrict__ dst,
                     int n) {
  int i = blockIdx.x * blockDim.x + threadIdx.x;
  if (i < n) dst[i] = (unsigned short)f2bf1(src[i]);
}

// ---------------------------------------------------------------------------
extern "C" void kernel_launch(void* const* d_in, const int* in_sizes, int n_in,
                              void* d_out, int out_size, void* d_ws, size_t ws_size,
                              hipStream_t stream) {
  const float* contents = (const float*)d_in[0];   // [16][NN][7]
  const int*   children = (const int*)  d_in[1];   // [15][NN][2]
  const float* Wu = (const float*)d_in[2];
  const float* bu = (const float*)d_in[3];
  const float* Wr = (const float*)d_in[4];
  const float* br = (const float*)d_in[5];
  const float* Wh = (const float*)d_in[6];
  const float* bh = (const float*)d_in[7];
  const float* Wz = (const float*)d_in[8];
  const float* bz = (const float*)d_in[9];
  float* out = (float*)d_out;

  unsigned char* ws = (unsigned char*)d_ws;
  float* hws = (float*)ws;                                     // [NN][64] f32
  unsigned short* WrB = (unsigned short*)(ws + (size_t)NN * HID * 4);
  unsigned short* WhB = WrB + 192 * 192;
  unsigned short* WzB = WhB + 64 * 192;

  wcvt<<<(192 * 192 + 255) / 256, 256, 0, stream>>>(Wr, WrB, 192 * 192);
  wcvt<<<(64 * 192 + 255) / 256, 256, 0, stream>>>(Wh, WhB, 64 * 192);
  wcvt<<<(256 * 256 + 255) / 256, 256, 0, stream>>>(Wz, WzB, 256 * 256);

  // level 15 leaves -> buffer for t=15 (odd -> ws)
  leaf_embed<<<(NN * HID) / 256, 256, 0, stream>>>(
      contents + (size_t)15 * NN * FEAT, Wu, bu, hws);

  // scan: t = 14 .. 0; buffer(t) = (t odd) ? ws : d_out  (final t=0 -> d_out)
  for (int t = 14; t >= 0; t--) {
    float* hO = (t & 1) ? hws : out;
    const float* hI = ((t + 1) & 1) ? hws : out;
    grnn_step<<<NN / 64, 128, 0, stream>>>(
        contents + (size_t)t * NN * FEAT, children + (size_t)t * NN * 2, hI,
        WrB, WhB, WzB, Wu, bu, br, bh, bz, hO);
  }
}
